// SketchMultiHeadAttention_26319559590403
// MI455X (gfx1250) — matrix operand order repaired and checked by reading, not compiled, not run on hardware
//
#include <hip/hip_runtime.h>

#define MHA_B  2
#define MHA_S  2048
#define MHA_D  1024
#define MHA_H  16
#define MHA_HD 64

typedef __bf16 bf16_t;
typedef __attribute__((ext_vector_type(16))) __bf16 v16bf;
typedef __attribute__((ext_vector_type(2)))  __bf16 bf2;
typedef __attribute__((ext_vector_type(8)))  float  v8f;

__device__ __forceinline__ v8f wmma_bf16(v16bf a, v16bf b, v8f c) {
  return __builtin_amdgcn_wmma_f32_16x16x32_bf16(
      /*neg_a=*/false, a, /*neg_b=*/false, b,
      /*c_mod=*/(short)0, c, /*reuse_a=*/false, /*reuse_b=*/false);
}

// A-fragment (16x32, 16-bit) K index for VGPR v, lane half (ISA 7.12.2)
__device__ __forceinline__ int a_k(int v, int half) {
  return ((v & 4) << 2) + 2 * (v & 3) + 8 * half;
}
// B-fragment (32x16, 16-bit) K index: same K order as the A-fragment
__device__ __forceinline__ int b_k(int v, int half) {
  return ((v & 4) << 2) + 2 * (v & 3) + 8 * half;
}

// ---------------------------------------------------------------------------
// Kernel 1: fused QKV projection.  C = x @ W + b, 64x128 block tile,
// 8 waves of 32x32 (2x2 WMMA).  z selects Q/K/V.  Q pre-scaled by 1/8.
// Q,K stored bf16 [b,h,s,hd]; V stored transposed bf16 [b,h,hd,s].
// ---------------------------------------------------------------------------
__global__ __launch_bounds__(256)
void qkv_gemm_kernel(const float* __restrict__ x,
                     const float* __restrict__ Wq, const float* __restrict__ bq,
                     const float* __restrict__ Wk, const float* __restrict__ bk,
                     const float* __restrict__ Wv, const float* __restrict__ bv,
                     bf16_t* __restrict__ Qbf, bf16_t* __restrict__ Kbf,
                     bf16_t* __restrict__ Vt) {
  const int proj = blockIdx.z;
  const float* W    = (proj == 0) ? Wq : (proj == 1) ? Wk : Wv;
  const float* bias = (proj == 0) ? bq : (proj == 1) ? bk : bv;

  __shared__ bf16_t Alds[64 * 32];    // 64 rows x 32 k
  __shared__ bf16_t Blds[128 * 32];   // transposed: [n][k]

  const int tid  = threadIdx.x;
  const int lane = tid & 31;
  const int wid  = tid >> 5;
  const int wm   = wid >> 2;          // 0..1
  const int wn   = wid & 3;           // 0..3
  const int half = lane >> 4;
  const int l16  = lane & 15;

  const int m0 = blockIdx.y * 64;
  const int n0 = blockIdx.x * 128;

  v8f vzero = {};
  v8f acc[2][2];
  for (int i = 0; i < 2; ++i)
    for (int j = 0; j < 2; ++j) acc[i][j] = vzero;

  const int ar = tid >> 2;            // A stage: row
  const int ac = (tid & 3) * 8;       // A stage: col base
  const int bk0 = tid >> 3;           // B stage: k row
  const int bc  = (tid & 7) * 16;     // B stage: col base

  for (int k0 = 0; k0 < MHA_D; k0 += 32) {
    { // stage A 64x32 (fp32 -> bf16)
      const float4* src = (const float4*)(x + (size_t)(m0 + ar) * MHA_D + k0 + ac);
      float4 f0 = src[0], f1 = src[1];
      bf16_t* dst = &Alds[ar * 32 + ac];
      dst[0] = (bf16_t)f0.x; dst[1] = (bf16_t)f0.y;
      dst[2] = (bf16_t)f0.z; dst[3] = (bf16_t)f0.w;
      dst[4] = (bf16_t)f1.x; dst[5] = (bf16_t)f1.y;
      dst[6] = (bf16_t)f1.z; dst[7] = (bf16_t)f1.w;
    }
    { // stage B 32x128 transposed (fp32 -> bf16)
      const float4* src = (const float4*)(W + (size_t)(k0 + bk0) * MHA_D + n0 + bc);
#pragma unroll
      for (int q = 0; q < 4; ++q) {
        float4 f = src[q];
        Blds[(bc + q * 4 + 0) * 32 + bk0] = (bf16_t)f.x;
        Blds[(bc + q * 4 + 1) * 32 + bk0] = (bf16_t)f.y;
        Blds[(bc + q * 4 + 2) * 32 + bk0] = (bf16_t)f.z;
        Blds[(bc + q * 4 + 3) * 32 + bk0] = (bf16_t)f.w;
      }
    }
    // prefetch next K-slice while this tile computes
    if (k0 + 32 < MHA_D) {
      __builtin_prefetch(x + (size_t)(m0 + ar) * MHA_D + k0 + 32 + ac, 0, 3);
      __builtin_prefetch(W + (size_t)(k0 + 32 + bk0) * MHA_D + n0 + bc, 0, 3);
    }
    __syncthreads();

    v16bf afrag[2], bfrag[2];
#pragma unroll
    for (int i = 0; i < 2; ++i) {
      int row = wm * 32 + i * 16 + l16;
#pragma unroll
      for (int v = 0; v < 8; ++v) {
        bf2 t = *(const bf2*)&Alds[row * 32 + a_k(v, half)];
        afrag[i][2 * v] = t.x; afrag[i][2 * v + 1] = t.y;
      }
    }
#pragma unroll
    for (int j = 0; j < 2; ++j) {
      int col = wn * 32 + j * 16 + l16;
#pragma unroll
      for (int v = 0; v < 8; ++v) {
        bf2 t = *(const bf2*)&Blds[col * 32 + b_k(v, half)];
        bfrag[j][2 * v] = t.x; bfrag[j][2 * v + 1] = t.y;
      }
    }
#pragma unroll
    for (int i = 0; i < 2; ++i)
#pragma unroll
      for (int j = 0; j < 2; ++j)
        acc[i][j] = wmma_bf16(afrag[i], bfrag[j], acc[i][j]);
    __syncthreads();
  }

  // ---- epilogue: proj branch hoisted OUTSIDE the unrolled store loops ----
  const float qscale = 0.125f;  // 1/sqrt(64), folded into Q
  if (proj == 0) {
#pragma unroll
    for (int i = 0; i < 2; ++i)
#pragma unroll
      for (int j = 0; j < 2; ++j)
#pragma unroll
        for (int v = 0; v < 8; ++v) {
          int m = m0 + wm * 32 + i * 16 + v + 8 * half;
          int n = n0 + wn * 32 + j * 16 + l16;
          float val = (acc[i][j][v] + bias[n]) * qscale;
          int b = m >> 11, s = m & 2047;
          int h = n >> 6,  d = n & 63;
          size_t bh = (size_t)(b * MHA_H + h);
          Qbf[(bh * MHA_S + s) * MHA_HD + d] = (bf16_t)val;
        }
  } else if (proj == 1) {
#pragma unroll
    for (int i = 0; i < 2; ++i)
#pragma unroll
      for (int j = 0; j < 2; ++j)
#pragma unroll
        for (int v = 0; v < 8; ++v) {
          int m = m0 + wm * 32 + i * 16 + v + 8 * half;
          int n = n0 + wn * 32 + j * 16 + l16;
          float val = acc[i][j][v] + bias[n];
          int b = m >> 11, s = m & 2047;
          int h = n >> 6,  d = n & 63;
          size_t bh = (size_t)(b * MHA_H + h);
          Kbf[(bh * MHA_S + s) * MHA_HD + d] = (bf16_t)val;
        }
  } else {
#pragma unroll
    for (int i = 0; i < 2; ++i)
#pragma unroll
      for (int j = 0; j < 2; ++j)
#pragma unroll
        for (int v = 0; v < 8; ++v) {
          int m = m0 + wm * 32 + i * 16 + v + 8 * half;
          int n = n0 + wn * 32 + j * 16 + l16;
          float val = acc[i][j][v] + bias[n];
          int b = m >> 11, s = m & 2047;
          int h = n >> 6,  d = n & 63;
          size_t bh = (size_t)(b * MHA_H + h);
          Vt[(bh * MHA_HD + d) * MHA_S + s] = (bf16_t)val;
        }
  }
}

// ---------------------------------------------------------------------------
// Kernel 2: flash attention. One wave per (16-query tile, head, batch).
// 32 keys / iteration: 4 WMMAs for scores, online softmax, 4 WMMAs for P*V.
// ---------------------------------------------------------------------------
__global__ __launch_bounds__(32)
void attn_kernel(const bf16_t* __restrict__ Qbf, const bf16_t* __restrict__ Kbf,
                 const bf16_t* __restrict__ Vt, const float* __restrict__ mask,
                 bf16_t* __restrict__ ctx) {
  __shared__ bf16_t Plds[16 * 32];

  const int lane = threadIdx.x & 31;
  const int half = lane >> 4;
  const int l16  = lane & 15;
  const int q0 = blockIdx.x * 16;
  const int h  = blockIdx.y;
  const int b  = blockIdx.z;
  const size_t bh = (size_t)(b * MHA_H + h);

  const bf16_t* Q = Qbf + (bh * MHA_S + q0) * MHA_HD;  // 16x64 tile
  const bf16_t* K = Kbf + bh * MHA_S * MHA_HD;         // [2048][64]
  const bf16_t* V = Vt  + bh * MHA_HD * MHA_S;         // [64][2048]
  const float*  mk = mask + (size_t)b * MHA_S;

  // Q as two 16x32 A-fragments (hd 0..31, 32..63)
  v16bf qa[2];
#pragma unroll
  for (int f = 0; f < 2; ++f)
#pragma unroll
    for (int v = 0; v < 8; ++v) {
      bf2 t = *(const bf2*)&Q[l16 * MHA_HD + f * 32 + a_k(v, half)];
      qa[f][2 * v] = t.x; qa[f][2 * v + 1] = t.y;
    }

  float mrow[8], lrow[8];
  v8f vzero = {};
  v8f accv[4];
#pragma unroll
  for (int v = 0; v < 8; ++v) { mrow[v] = -1e30f; lrow[v] = 0.0f; }
#pragma unroll
  for (int j = 0; j < 4; ++j) accv[j] = vzero;

  for (int kk = 0; kk < MHA_S; kk += 32) {
    // --- scores: two 16x16 tiles over keys kk..kk+31 ---
    v8f sc[2];
#pragma unroll
    for (int t = 0; t < 2; ++t) {
      int key = kk + t * 16 + l16;
      v16bf kb0, kb1;
#pragma unroll
      for (int v = 0; v < 8; ++v) {
        int k = b_k(v, half);
        bf2 u0 = *(const bf2*)&K[(size_t)key * MHA_HD + k];
        bf2 u1 = *(const bf2*)&K[(size_t)key * MHA_HD + 32 + k];
        kb0[2 * v] = u0.x; kb0[2 * v + 1] = u0.y;
        kb1[2 * v] = u1.x; kb1[2 * v + 1] = u1.y;
      }
      v8f s = vzero;
      s = wmma_bf16(qa[0], kb0, s);
      s = wmma_bf16(qa[1], kb1, s);
      float mval = mk[key];
#pragma unroll
      for (int v = 0; v < 8; ++v) s[v] = s[v] + mval;
      sc[t] = s;
    }

    // --- online softmax over these 32 columns ---
#pragma unroll
    for (int v = 0; v < 8; ++v) {
      float mx = fmaxf(sc[0][v], sc[1][v]);
#pragma unroll
      for (int off = 1; off < 16; off <<= 1)
        mx = fmaxf(mx, __shfl_xor(mx, off, 32));
      float mnew = fmaxf(mrow[v], mx);
      float corr = __expf(mrow[v] - mnew);
      float p0 = __expf(sc[0][v] - mnew);
      float p1 = __expf(sc[1][v] - mnew);
      float ps = p0 + p1;
#pragma unroll
      for (int off = 1; off < 16; off <<= 1)
        ps += __shfl_xor(ps, off, 32);
      lrow[v] = lrow[v] * corr + ps;
      mrow[v] = mnew;
#pragma unroll
      for (int j = 0; j < 4; ++j) accv[j][v] = accv[j][v] * corr;
      int row = v + 8 * half;
      Plds[row * 32 + l16]      = (bf16_t)p0;
      Plds[row * 32 + 16 + l16] = (bf16_t)p1;
    }
    __syncthreads();  // single-wave WG: compiler fence + DS ordering

    // --- re-layout P into A-fragment (16 rows x 32 keys) ---
    v16bf pa;
#pragma unroll
    for (int v = 0; v < 8; ++v) {
      bf2 t = *(const bf2*)&Plds[l16 * 32 + a_k(v, half)];
      pa[2 * v] = t.x; pa[2 * v + 1] = t.y;
    }
    __syncthreads();

    // --- P * V : 4 d-tiles of 16 ---
#pragma unroll
    for (int j = 0; j < 4; ++j) {
      v16bf vb;
      int d = j * 16 + l16;
#pragma unroll
      for (int v = 0; v < 8; ++v) {
        bf2 t = *(const bf2*)&V[(size_t)d * MHA_S + kk + b_k(v, half)];
        vb[2 * v] = t.x; vb[2 * v + 1] = t.y;
      }
      accv[j] = wmma_bf16(pa, vb, accv[j]);
    }
  }

  // finalize: ctx[b][s][h*64+d] = acc / l   (bf16 for output GEMM)
#pragma unroll
  for (int j = 0; j < 4; ++j)
#pragma unroll
    for (int v = 0; v < 8; ++v) {
      int row = v + 8 * half;
      int d = j * 16 + l16;
      float val = accv[j][v] / lrow[v];
      int s = q0 + row;
      ctx[((size_t)b * MHA_S + s) * MHA_D + h * MHA_HD + d] = (bf16_t)val;
    }
}

// ---------------------------------------------------------------------------
// Kernel 3: output projection + bias + residual (fp32 out to workspace)
// ---------------------------------------------------------------------------
__global__ __launch_bounds__(256)
void out_gemm_kernel(const bf16_t* __restrict__ ctx, const float* __restrict__ Wo,
                     const float* __restrict__ bo, const float* __restrict__ x,
                     float* __restrict__ resid) {
  __shared__ bf16_t Alds[64 * 32];
  __shared__ bf16_t Blds[128 * 32];

  const int tid  = threadIdx.x;
  const int lane = tid & 31;
  const int wid  = tid >> 5;
  const int wm   = wid >> 2;
  const int wn   = wid & 3;
  const int half = lane >> 4;
  const int l16  = lane & 15;

  const int m0 = blockIdx.y * 64;
  const int n0 = blockIdx.x * 128;

  v8f vzero = {};
  v8f acc[2][2];
  for (int i = 0; i < 2; ++i)
    for (int j = 0; j < 2; ++j) acc[i][j] = vzero;

  const int ar = tid >> 2;
  const int ac = (tid & 3) * 8;
  const int bk0 = tid >> 3;
  const int bc  = (tid & 7) * 16;

  for (int k0 = 0; k0 < MHA_D; k0 += 32) {
    { // stage A (already bf16): 16-byte vector copy
      *(uint4*)&Alds[ar * 32 + ac] =
          *(const uint4*)(ctx + (size_t)(m0 + ar) * MHA_D + k0 + ac);
    }
    { // stage Wo transposed
      const float4* src = (const float4*)(Wo + (size_t)(k0 + bk0) * MHA_D + n0 + bc);
#pragma unroll
      for (int q = 0; q < 4; ++q) {
        float4 f = src[q];
        Blds[(bc + q * 4 + 0) * 32 + bk0] = (bf16_t)f.x;
        Blds[(bc + q * 4 + 1) * 32 + bk0] = (bf16_t)f.y;
        Blds[(bc + q * 4 + 2) * 32 + bk0] = (bf16_t)f.z;
        Blds[(bc + q * 4 + 3) * 32 + bk0] = (bf16_t)f.w;
      }
    }
    if (k0 + 32 < MHA_D) {
      __builtin_prefetch(ctx + (size_t)(m0 + ar) * MHA_D + k0 + 32 + ac, 0, 3);
      __builtin_prefetch(Wo + (size_t)(k0 + 32 + bk0) * MHA_D + n0 + bc, 0, 3);
    }
    __syncthreads();

    v16bf afrag[2], bfrag[2];
#pragma unroll
    for (int i = 0; i < 2; ++i) {
      int row = wm * 32 + i * 16 + l16;
#pragma unroll
      for (int v = 0; v < 8; ++v) {
        bf2 t = *(const bf2*)&Alds[row * 32 + a_k(v, half)];
        afrag[i][2 * v] = t.x; afrag[i][2 * v + 1] = t.y;
      }
    }
#pragma unroll
    for (int j = 0; j < 2; ++j) {
      int col = wn * 32 + j * 16 + l16;
#pragma unroll
      for (int v = 0; v < 8; ++v) {
        bf2 t = *(const bf2*)&Blds[col * 32 + b_k(v, half)];
        bfrag[j][2 * v] = t.x; bfrag[j][2 * v + 1] = t.y;
      }
    }
#pragma unroll
    for (int i = 0; i < 2; ++i)
#pragma unroll
      for (int j = 0; j < 2; ++j)
        acc[i][j] = wmma_bf16(afrag[i], bfrag[j], acc[i][j]);
    __syncthreads();
  }

#pragma unroll
  for (int i = 0; i < 2; ++i)
#pragma unroll
    for (int j = 0; j < 2; ++j)
#pragma unroll
      for (int v = 0; v < 8; ++v) {
        int m = m0 + wm * 32 + i * 16 + v + 8 * half;
        int n = n0 + wn * 32 + j * 16 + l16;
        size_t idx = (size_t)m * MHA_D + n;
        resid[idx] = acc[i][j][v] + bo[n] + x[idx];
      }
}

// ---------------------------------------------------------------------------
// Kernel 4: row-wise LayerNorm -> final output
// ---------------------------------------------------------------------------
__global__ __launch_bounds__(256)
void ln_kernel(const float* __restrict__ resid, const float* __restrict__ gamma,
               const float* __restrict__ beta, float* __restrict__ out) {
  __shared__ float s1[8], s2[8];
  const int row = blockIdx.x;
  const int tid = threadIdx.x;
  const float* r = resid + (size_t)row * MHA_D;

  float vals[4];
  float sum = 0.0f, sumsq = 0.0f;
#pragma unroll
  for (int i = 0; i < 4; ++i) {
    float v = r[tid + i * 256];
    vals[i] = v; sum += v; sumsq += v * v;
  }
  for (int off = 1; off < 32; off <<= 1) {
    sum   += __shfl_xor(sum, off, 32);
    sumsq += __shfl_xor(sumsq, off, 32);
  }
  int wid = tid >> 5, lane = tid & 31;
  if (lane == 0) { s1[wid] = sum; s2[wid] = sumsq; }
  __syncthreads();
  if (wid == 0) {
    float a = (lane < 8) ? s1[lane] : 0.0f;
    float c = (lane < 8) ? s2[lane] : 0.0f;
    for (int off = 1; off < 8; off <<= 1) {
      a += __shfl_xor(a, off, 32);
      c += __shfl_xor(c, off, 32);
    }
    if (lane == 0) { s1[0] = a; s2[0] = c; }
  }
  __syncthreads();
  float mean = s1[0] * (1.0f / MHA_D);
  float var  = s2[0] * (1.0f / MHA_D) - mean * mean;
  float inv  = rsqrtf(var + 1e-5f);
#pragma unroll
  for (int i = 0; i < 4; ++i) {
    int c = tid + i * 256;
    out[(size_t)row * MHA_D + c] = (vals[i] - mean) * inv * gamma[c] + beta[c];
  }
}

// ---------------------------------------------------------------------------
extern "C" void kernel_launch(void* const* d_in, const int* in_sizes, int n_in,
                              void* d_out, int out_size, void* d_ws, size_t ws_size,
                              hipStream_t stream) {
  const float* x    = (const float*)d_in[0];
  const float* mask = (const float*)d_in[1];
  const float* Wq = (const float*)d_in[2];
  const float* bq = (const float*)d_in[3];
  const float* Wk = (const float*)d_in[4];
  const float* bk = (const float*)d_in[5];
  const float* Wv = (const float*)d_in[6];
  const float* bv = (const float*)d_in[7];
  const float* Wo = (const float*)d_in[8];
  const float* bo = (const float*)d_in[9];
  const float* g  = (const float*)d_in[10];
  const float* be = (const float*)d_in[11];
  float* out = (float*)d_out;

  char* ws = (char*)d_ws;
  const size_t MB8 = (size_t)8 << 20;
  bf16_t* Qbf   = (bf16_t*)(ws);               //  8 MB
  bf16_t* Kbf   = (bf16_t*)(ws + 1 * MB8);     //  8 MB
  bf16_t* Vt    = (bf16_t*)(ws + 2 * MB8);     //  8 MB
  bf16_t* ctx   = (bf16_t*)(ws + 3 * MB8);     //  8 MB
  float*  resid = (float*) (ws + 4 * MB8);     // 16 MB  (total 48 MB)

  qkv_gemm_kernel<<<dim3(MHA_D / 128, (MHA_B * MHA_S) / 64, 3), 256, 0, stream>>>(
      x, Wq, bq, Wk, bk, Wv, bv, Qbf, Kbf, Vt);
  attn_kernel<<<dim3(MHA_S / 16, MHA_H, MHA_B), 32, 0, stream>>>(
      Qbf, Kbf, Vt, mask, ctx);
  out_gemm_kernel<<<dim3(MHA_D / 128, (MHA_B * MHA_S) / 64), 256, 0, stream>>>(
      ctx, Wo, bo, x, resid);
  ln_kernel<<<MHA_B * MHA_S, 256, 0, stream>>>(resid, g, be, out);
}
